// CustomNetwork_46660524704376
// MI455X (gfx1250) — matrix operand order repaired and checked by reading, not compiled, not run on hardware
//
#include <hip/hip_runtime.h>

typedef unsigned int u32;
typedef unsigned short u16;
typedef __attribute__((ext_vector_type(4)))  u32    u32x4;
typedef __attribute__((ext_vector_type(16))) __bf16 v16bf;
typedef __attribute__((ext_vector_type(8)))  float  v8f;
typedef int v4i __attribute__((vector_size(16)));

static constexpr int D = 4096;        // width (= batch = K for every layer)
static constexpr int NELEM = D * D;   // 16.7M

#if __has_builtin(__builtin_amdgcn_global_load_async_to_lds_b128)
#define HAS_ASYNC_LDS 1
#endif

static __device__ __forceinline__ u16 f32_to_bf16(float f) {
    u32 u = __float_as_uint(f);
    u = (u + 0x7FFFu + ((u >> 16) & 1u)) >> 16;   // round-to-nearest-even
    return (u16)u;
}

// 16-byte global -> LDS copy; async DMA (ASYNCcnt) when available.
static __device__ __forceinline__ void copy16_g2l(const u16* g, u16* l) {
#ifdef HAS_ASYNC_LDS
    __builtin_amdgcn_global_load_async_to_lds_b128(
        (__attribute__((address_space(1))) v4i*)(void*)const_cast<u16*>(g),
        (__attribute__((address_space(3))) v4i*)(void*)l, 0, 0);
#else
    *(u32x4*)l = *(const u32x4*)g;   // sync fallback: global->VGPR->LDS
#endif
}

template <int N>
static __device__ __forceinline__ void wait_async_n() {
#ifdef HAS_ASYNC_LDS
#if __has_builtin(__builtin_amdgcn_s_wait_asynccnt)
    __builtin_amdgcn_s_wait_asynccnt((unsigned short)N);
#else
    asm volatile("s_wait_asynccnt %0" ::"i"(N) : "memory");
#endif
#endif
}

// ---------------- W: f32 [K][N] -> bf16 [N][K] (transpose + convert) ----------------
__global__ void transpose_convert_w(const float* __restrict__ W, u16* __restrict__ Wt) {
    __shared__ float tile[32][33];
    const int n0 = blockIdx.x * 32;
    const int k0 = blockIdx.y * 32;
    const int tx = threadIdx.x;       // 0..31
    const int ty = threadIdx.y;       // 0..7
    for (int j = ty; j < 32; j += 8)
        tile[j][tx] = W[(size_t)(k0 + j) * D + n0 + tx];
    __syncthreads();
    for (int j = ty; j < 32; j += 8)
        Wt[(size_t)(n0 + j) * D + k0 + tx] = f32_to_bf16(tile[tx][j]);
}

// ---------------- f32 -> bf16 elementwise ----------------
__global__ void cvt_bf16(const float* __restrict__ in, u16* __restrict__ out) {
    const int i = blockIdx.x * 256 + threadIdx.x;
    out[i] = f32_to_bf16(in[i]);
}

// ---------------- bf16 WMMA GEMM: Y[M][N] = Hb[M][K] x Wt[N][K]^T + bias[N] -------
// Block tile 128(M) x 256(N), KC=32, double-buffered LDS fed by async LDS-DMA.
// 8 waves as 2(M) x 4(N); wave tile 64x64 = 4x4 WMMA tiles.
__global__ __launch_bounds__(256) void gemm_bf16(const u16* __restrict__ A,   // [M][K] bf16
                                                 const u16* __restrict__ Bt,  // [N][K] bf16
                                                 const float* __restrict__ bias,
                                                 float* __restrict__ Y) {
    __shared__ __align__(16) u16 As[2][128 * 32];   // 2 x 8 KB
    __shared__ __align__(16) u16 Bs[2][256 * 32];   // 2 x 16 KB

    const int tid  = threadIdx.x;
    const int lane = tid & 31;
    const int wave = tid >> 5;       // 0..7
    const int wm   = wave & 1;       // 2 waves along M -> 64 rows each
    const int wn   = wave >> 1;      // 4 waves along N -> 64 cols each
    const int m0   = blockIdx.x * 128;
    const int n0   = blockIdx.y * 256;
    const int r16  = lane & 15;
    const int hi   = lane >> 4;      // 0 or 1

    v8f acc[4][4];
    for (int nt = 0; nt < 4; ++nt) {
        const float bv = bias[n0 + wn * 64 + nt * 16 + r16];
        for (int mt = 0; mt < 4; ++mt)
            for (int r = 0; r < 8; ++r) acc[mt][nt][r] = bv;
    }

    // staging map: 16B units; A tile 128x32 = 512 units (2/thread), B tile 256x32 = 1024 (4/thread)
    const int uRow = tid >> 2;          // 0..63
    const int uSeg = (tid & 3) * 8;     // 0,8,16,24 (u16 elements)
    const int kA   = hi ? 8 : 0;        // A frag K base (interleaved layout)
    const int kB   = hi ? 8 : 0;        // B frag K base (interleaved layout)

    constexpr int NIT = D / 32;         // 128 k-chunks

    auto stage = [&](int buf, int k0) {
        u16* as = &As[buf][0];
        u16* bs = &Bs[buf][0];
#pragma unroll
        for (int t = 0; t < 2; ++t) {
            const int row = uRow + t * 64;
            copy16_g2l(A + (size_t)(m0 + row) * D + k0 + uSeg, as + row * 32 + uSeg);
        }
#pragma unroll
        for (int t = 0; t < 4; ++t) {
            const int row = uRow + t * 64;
            copy16_g2l(Bt + (size_t)(n0 + row) * D + k0 + uSeg, bs + row * 32 + uSeg);
        }
    };

    stage(0, 0);
    for (int i = 0; i < NIT; ++i) {
        const int cur = i & 1;
        if (i + 1 < NIT) {
            stage(cur ^ 1, (i + 1) * 32);
            wait_async_n<6>();          // current chunk's 6 DMAs (in-order) have landed
        } else {
            wait_async_n<0>();
        }
        __syncthreads();                // all waves' current-buffer data visible

        const u16* as = &As[cur][0];
        const u16* bs = &Bs[cur][0];
        v16bf fa[4], fb[4];
#pragma unroll
        for (int mt = 0; mt < 4; ++mt) {
            const int rl = wm * 64 + mt * 16 + r16;
            union { u32x4 u[2]; v16bf v; } t;
            t.u[0] = *(const u32x4*)(as + rl * 32 + kA);
            t.u[1] = *(const u32x4*)(as + rl * 32 + kA + 16);
            fa[mt] = t.v;
        }
#pragma unroll
        for (int nt = 0; nt < 4; ++nt) {
            const int cl = wn * 64 + nt * 16 + r16;
            union { u32x4 u[2]; v16bf v; } t;
            t.u[0] = *(const u32x4*)(bs + cl * 32 + kB);
            t.u[1] = *(const u32x4*)(bs + cl * 32 + kB + 16);
            fb[nt] = t.v;
        }
#pragma unroll
        for (int mt = 0; mt < 4; ++mt)
#pragma unroll
            for (int nt = 0; nt < 4; ++nt)
                acc[mt][nt] = __builtin_amdgcn_wmma_f32_16x16x32_bf16(
                    false, fa[mt], false, fb[nt], (short)0, acc[mt][nt], false, false);

        __syncthreads();                // everyone done reading cur before it is restaged
    }

#pragma unroll
    for (int mt = 0; mt < 4; ++mt)
#pragma unroll
        for (int nt = 0; nt < 4; ++nt) {
            const int col   = n0 + wn * 64 + nt * 16 + r16;
            const int rbase = m0 + wm * 64 + mt * 16 + hi * 8;
#pragma unroll
            for (int r = 0; r < 8; ++r)
                Y[(size_t)(rbase + r) * D + col] = acc[mt][nt][r];
        }
}

// ---------------- column stats (mean/var over batch dim), two-stage ----------------
__global__ void col_stats_partial(const float* __restrict__ Y,
                                  float* __restrict__ Ps, float* __restrict__ Ps2) {
    const int col = blockIdx.x * 256 + threadIdx.x;
    const int r0  = blockIdx.y * 256;
    float s = 0.f, s2 = 0.f;
    for (int r = 0; r < 256; ++r) {
        const float v = Y[(size_t)(r0 + r) * D + col];
        s += v; s2 += v * v;
    }
    Ps [blockIdx.y * D + col] = s;
    Ps2[blockIdx.y * D + col] = s2;
}

__global__ void col_stats_final(const float* __restrict__ Ps, const float* __restrict__ Ps2,
                                float* __restrict__ mu, float* __restrict__ rstd) {
    const int col = blockIdx.x * 256 + threadIdx.x;
    float s = 0.f, s2 = 0.f;
    for (int c = 0; c < 16; ++c) { s += Ps[c * D + col]; s2 += Ps2[c * D + col]; }
    const float m = s * (1.0f / D);
    float var = s2 * (1.0f / D) - m * m;
    var = fmaxf(var, 0.f);
    mu[col]   = m;
    rstd[col] = rsqrtf(var + 1e-5f);
}

// ---------------- fused BN + ReLU; emit bf16 activations and/or f32 ----------------
__global__ void bn_relu(const float* __restrict__ Y,
                        const float* __restrict__ mu, const float* __restrict__ rstd,
                        const float* __restrict__ gamma, const float* __restrict__ beta,
                        u16* __restrict__ Hb, float* __restrict__ Yout,
                        int write_bf16, int write_f32) {
    const int i   = blockIdx.x * 256 + threadIdx.x;
    const int col = i & (D - 1);
    float v = (Y[i] - mu[col]) * rstd[col] * gamma[col] + beta[col];
    v = fmaxf(v, 0.f);
    if (write_bf16) Hb[i] = f32_to_bf16(v);
    if (write_f32)  Yout[i] = v;
}

// ---------------- softmax over batch dim (per column), two-stage ----------------
__global__ void col_max_partial(const float* __restrict__ Y, float* __restrict__ Pm) {
    const int col = blockIdx.x * 256 + threadIdx.x;
    const int r0  = blockIdx.y * 256;
    float m = -3.402823466e38f;
    for (int r = 0; r < 256; ++r) m = fmaxf(m, Y[(size_t)(r0 + r) * D + col]);
    Pm[blockIdx.y * D + col] = m;
}
__global__ void col_max_final(const float* __restrict__ Pm, float* __restrict__ cmax) {
    const int col = blockIdx.x * 256 + threadIdx.x;
    float m = -3.402823466e38f;
    for (int c = 0; c < 16; ++c) m = fmaxf(m, Pm[c * D + col]);
    cmax[col] = m;
}
__global__ void col_expsum_partial(const float* __restrict__ Y, const float* __restrict__ cmax,
                                   float* __restrict__ Ps) {
    const int col = blockIdx.x * 256 + threadIdx.x;
    const int r0  = blockIdx.y * 256;
    const float cm = cmax[col];
    float s = 0.f;
    for (int r = 0; r < 256; ++r) s += __expf(Y[(size_t)(r0 + r) * D + col] - cm);
    Ps[blockIdx.y * D + col] = s;
}
__global__ void col_expsum_final(const float* __restrict__ Ps, float* __restrict__ csum) {
    const int col = blockIdx.x * 256 + threadIdx.x;
    float s = 0.f;
    for (int c = 0; c < 16; ++c) s += Ps[c * D + col];
    csum[col] = s;
}
__global__ void softmax_write(float* __restrict__ Y, const float* __restrict__ cmax,
                              const float* __restrict__ csum) {
    const int i   = blockIdx.x * 256 + threadIdx.x;
    const int col = i & (D - 1);
    Y[i] = __expf(Y[i] - cmax[col]) / csum[col];
}

extern "C" void kernel_launch(void* const* d_in, const int* in_sizes, int n_in,
                              void* d_out, int out_size, void* d_ws, size_t ws_size,
                              hipStream_t stream) {
    (void)in_sizes; (void)n_in; (void)out_size; (void)ws_size;
    const float* x     = (const float*)d_in[0];
    const float* W     = (const float*)d_in[1];   // [4][D][D]
    const float* b     = (const float*)d_in[2];   // [4][D]
    const float* gamma = (const float*)d_in[3];   // [4][D]
    const float* beta  = (const float*)d_in[4];   // [4][D]
    float* out = (float*)d_out;                   // doubles as f32 Y scratch (same size)

    // Workspace layout (~66 MB): Wt bf16 (32MB, one layer at a time) | Hb bf16 (32MB) | partials
    char* ws   = (char*)d_ws;
    u16*  Wt   = (u16*)ws;
    u16*  Hb   = (u16*)(ws + (size_t)NELEM * 2);
    float* Ps   = (float*)(ws + (size_t)NELEM * 4);
    float* Ps2  = Ps  + 16 * D;
    float* mu   = Ps2 + 16 * D;
    float* rstd = mu  + D;
    float* cmax = rstd + D;
    float* csum = cmax + D;

    const int EW_GRID = NELEM / 256;              // 65536 blocks of 256

    cvt_bf16<<<EW_GRID, 256, 0, stream>>>(x, Hb);

    for (int l = 0; l < 4; ++l) {
        transpose_convert_w<<<dim3(D / 32, D / 32), dim3(32, 8), 0, stream>>>(
            W + (size_t)l * NELEM, Wt);
        gemm_bf16<<<dim3(D / 128, D / 256), 256, 0, stream>>>(Hb, Wt, b + l * D, out);
        col_stats_partial<<<dim3(16, 16), 256, 0, stream>>>(out, Ps, Ps2);
        col_stats_final<<<16, 256, 0, stream>>>(Ps, Ps2, mu, rstd);
        const int wb16 = (l < 3) ? 1 : 0;
        const int wf32 = (l == 3) ? 1 : 0;
        bn_relu<<<EW_GRID, 256, 0, stream>>>(out, mu, rstd, gamma + l * D, beta + l * D,
                                             Hb, out, wb16, wf32);
    }

    col_max_partial<<<dim3(16, 16), 256, 0, stream>>>(out, Ps);
    col_max_final<<<16, 256, 0, stream>>>(Ps, cmax);
    col_expsum_partial<<<dim3(16, 16), 256, 0, stream>>>(out, cmax, Ps2);
    col_expsum_final<<<16, 256, 0, stream>>>(Ps2, csum);
    softmax_write<<<EW_GRID, 256, 0, stream>>>(out, cmax, csum);
}
